// SimpleTransformerCUDA_4449586119383
// MI455X (gfx1250) — matrix operand order repaired and checked by reading, not compiled, not run on hardware
//
#include <hip/hip_runtime.h>
#include <math.h>

// Problem constants (fixed by the reference)
#define D_MODEL 1024
#define HEADS   16
#define HEAD_DIM 64
#define BATCH   4
#define SEQ     2048
#define M_TOK   (BATCH * SEQ)   // 8192 tokens

typedef __attribute__((ext_vector_type(16))) __bf16        v16bf;
typedef __attribute__((ext_vector_type(8)))  float         v8f;
typedef __attribute__((ext_vector_type(4)))  unsigned int  u32x4;

union FragU { v16bf v; u32x4 q[2]; };

// ---------------------------------------------------------------------------
// WMMA fragment loaders (bf16, 16x16x32 shapes, wave32 layouts per CDNA5 ISA)
// ---------------------------------------------------------------------------

// A-matrix 16x32 (MxK), source row-major with row stride `ld` elements.
// lanes 0-15: row=lane, K=0..7 then 16..23 ; lanes 16-31: row=lane-16, K=8..15 then 24..31
__device__ __forceinline__ v16bf load_a_frag(const __bf16* base, int ld) {
  const int lane = threadIdx.x & 31;
  const int row  = lane & 15;
  const int kb   = (lane < 16) ? 0 : 8;
  const __bf16* p = base + (size_t)row * ld;
  FragU r;
  r.q[0] = *(const u32x4*)(p + kb);
  r.q[1] = *(const u32x4*)(p + kb + 16);
  return r.v;
}

// B-matrix 32x16 (KxN) supplied as B^T in memory: Bt[n][k], row stride `ld`.
// lanes 0-15: col=lane, K=0..7 then 16..23 ; lanes 16-31: col=lane-16, K=8..15 then 24..31
__device__ __forceinline__ v16bf load_bt_frag(const __bf16* base, int ld) {
  const int lane = threadIdx.x & 31;
  const int col  = lane & 15;
  const int kb   = (lane < 16) ? 0 : 8;
  const __bf16* p = base + (size_t)col * ld + kb;
  FragU r;
  r.q[0] = *(const u32x4*)(p);
  r.q[1] = *(const u32x4*)(p + 16);
  return r.v;
}

__device__ __forceinline__ v8f wmma_bf16(v16bf a, v16bf b, v8f c) {
  return __builtin_amdgcn_wmma_f32_16x16x32_bf16(false, a, false, b, (short)0, c,
                                                 false, false);
}

// ---------------------------------------------------------------------------
// Conversion kernels
// ---------------------------------------------------------------------------
__global__ __launch_bounds__(256) void cvt_f32_to_bf16(const float* __restrict__ in,
                                                       __bf16* __restrict__ out, int n) {
  int i = blockIdx.x * 256 + threadIdx.x;
  if (i < n) out[i] = (__bf16)in[i];
}

// W [k][n] f32 -> Wt [n][k] bf16  (1024x1024)
__global__ __launch_bounds__(256) void transpose_w_bf16(const float* __restrict__ W,
                                                        __bf16* __restrict__ Wt) {
  int idx = blockIdx.x * 256 + threadIdx.x;      // 0 .. 1M-1
  int n = idx & (D_MODEL - 1);
  int k = idx >> 10;
  Wt[(size_t)n * D_MODEL + k] = (__bf16)W[(size_t)k * D_MODEL + n];
}

// ---------------------------------------------------------------------------
// bf16 WMMA GEMM:  C[M,N] = A[M,K] * Bt[N,K]^T
// block = 4 waves; wave tile = 32(M) x 64(N)  (2 A-frags, B-frags reused x2);
// block tile = 128(M) x 64(N).
// MODE 0: write bf16 row-major   (Q, K projections)
// MODE 1: write bf16 transposed per head -> Vt[b][h][hd][s]  (V projection)
// MODE 2: write f32 row-major    (final output projection)
// ---------------------------------------------------------------------------
template <int MODE>
__global__ __launch_bounds__(128)
void gemm_wmma_bf16(const __bf16* __restrict__ A, const __bf16* __restrict__ Bt,
                    void* __restrict__ Cout, int M, int N, int Kd) {
  const int warp = threadIdx.x >> 5;
  const int m0 = blockIdx.y * 128 + warp * 32;
  const int n0 = blockIdx.x * 64;

  v8f acc[2][4];
#pragma unroll
  for (int r = 0; r < 2; ++r)
#pragma unroll
    for (int t = 0; t < 4; ++t) acc[r][t] = v8f{};

  const __bf16* arow0 = A + (size_t)m0 * Kd;
  const __bf16* arow1 = A + (size_t)(m0 + 16) * Kd;
  for (int k = 0; k < Kd; k += 32) {
    v16bf a0 = load_a_frag(arow0 + k, Kd);
    v16bf a1 = load_a_frag(arow1 + k, Kd);
#pragma unroll
    for (int t = 0; t < 4; ++t) {
      v16bf bf = load_bt_frag(Bt + (size_t)(n0 + t * 16) * Kd + k, Kd);
      acc[0][t] = wmma_bf16(a0, bf, acc[0][t]);
      acc[1][t] = wmma_bf16(a1, bf, acc[1][t]);
    }
  }

  const int lane = threadIdx.x & 31;
  const int col  = lane & 15;
  const int rb   = (lane < 16) ? 0 : 8;

#pragma unroll
  for (int r = 0; r < 2; ++r) {
#pragma unroll
    for (int t = 0; t < 4; ++t) {
#pragma unroll
      for (int v = 0; v < 8; ++v) {
        const int m = m0 + r * 16 + rb + v;
        const int n = n0 + t * 16 + col;
        if (MODE == 0) {
          ((__bf16*)Cout)[(size_t)m * N + n] = (__bf16)acc[r][t][v];
        } else if (MODE == 2) {
          ((float*)Cout)[(size_t)m * N + n] = acc[r][t][v];
        } else {  // MODE 1: Vt[b][h][hd][s]
          const int b  = m / SEQ;
          const int s  = m & (SEQ - 1);
          const int h  = n / HEAD_DIM;
          const int hd = n & (HEAD_DIM - 1);
          ((__bf16*)Cout)[((size_t)(b * HEADS + h) * HEAD_DIM + hd) * SEQ + s] =
              (__bf16)acc[r][t][v];
        }
      }
    }
  }
}

// ---------------------------------------------------------------------------
// Fused attention (flash-attention style, online softmax).
// One wave per (b, h, 16-row q tile). Block = 4 waves. 32 keys per step.
// Q,K: bf16 [B*S, D] row-major. Vt: bf16 [B][H][HEAD_DIM][S]. merged: bf16 [B*S, D].
// ---------------------------------------------------------------------------
__global__ __launch_bounds__(128)
void attn_wmma(const __bf16* __restrict__ Q, const __bf16* __restrict__ Kp,
               const __bf16* __restrict__ Vt, __bf16* __restrict__ merged) {
  __shared__ __align__(16) __bf16 pTile[4][16 * 32];

  const int warp = threadIdx.x >> 5;
  const int lane = threadIdx.x & 31;
  const int col  = lane & 15;
  const int rb   = (lane < 16) ? 0 : 8;

  const int waveId = blockIdx.x * 4 + warp;
  const int qt = waveId & (SEQ / 16 - 1);           // 0..127
  const int h  = (waveId >> 7) & (HEADS - 1);       // 0..15
  const int b  = waveId >> 11;                      // 0..3
  const int q0 = qt * 16;

  __bf16* pT = pTile[warp];

  // Q fragments for this tile (head_dim 64 = 2 K-steps of 32)
  const __bf16* qbase = Q + ((size_t)(b * SEQ + q0)) * D_MODEL + h * HEAD_DIM;
  const v16bf aq0 = load_a_frag(qbase, D_MODEL);
  const v16bf aq1 = load_a_frag(qbase + 32, D_MODEL);

  float m_i[8], l_i[8];
#pragma unroll
  for (int v = 0; v < 8; ++v) { m_i[v] = -INFINITY; l_i[v] = 0.0f; }
  v8f o[4] = {v8f{}, v8f{}, v8f{}, v8f{}};

  const float scale = 0.125f;  // 1/sqrt(64)
  const __bf16* vbase = Vt + (size_t)(b * HEADS + h) * HEAD_DIM * SEQ;

  for (int jt = 0; jt < SEQ / 32; ++jt) {
    const __bf16* kb0 =
        Kp + ((size_t)(b * SEQ + jt * 32)) * D_MODEL + h * HEAD_DIM;
    const __bf16* kb1 = kb0 + (size_t)16 * D_MODEL;

    // scores: two 16x16 tiles (keys jt*32..+15 and +16..+31)
    v8f s0 = v8f{}, s1 = v8f{};
    s0 = wmma_bf16(aq0, load_bt_frag(kb0, D_MODEL), s0);
    s0 = wmma_bf16(aq1, load_bt_frag(kb0 + 32, D_MODEL), s0);
    s1 = wmma_bf16(aq0, load_bt_frag(kb1, D_MODEL), s1);
    s1 = wmma_bf16(aq1, load_bt_frag(kb1 + 32, D_MODEL), s1);

    float fac[8];
#pragma unroll
    for (int v = 0; v < 8; ++v) {
      s0[v] *= scale;
      s1[v] *= scale;
      // row max over 32 keys: local pair max, then reduce over the 16-lane half
      float t = fmaxf(s0[v], s1[v]);
      t = fmaxf(t, __shfl_xor(t, 1, 32));
      t = fmaxf(t, __shfl_xor(t, 2, 32));
      t = fmaxf(t, __shfl_xor(t, 4, 32));
      t = fmaxf(t, __shfl_xor(t, 8, 32));
      const float m_new = fmaxf(m_i[v], t);
      fac[v] = __expf(m_i[v] - m_new);
      m_i[v] = m_new;
      const float p0 = __expf(s0[v] - m_new);
      const float p1 = __expf(s1[v] - m_new);
      s0[v] = p0;
      s1[v] = p1;
      float rs = p0 + p1;
      rs += __shfl_xor(rs, 1, 32);
      rs += __shfl_xor(rs, 2, 32);
      rs += __shfl_xor(rs, 4, 32);
      rs += __shfl_xor(rs, 8, 32);
      l_i[v] = l_i[v] * fac[v] + rs;
    }

    // rescale accumulators
#pragma unroll
    for (int t = 0; t < 4; ++t)
#pragma unroll
      for (int v = 0; v < 8; ++v) o[t][v] *= fac[v];

    // stage P (16x32 bf16, row-major) through this wave's private LDS tile
#pragma unroll
    for (int v = 0; v < 8; ++v) {
      pT[(rb + v) * 32 + col]      = (__bf16)s0[v];
      pT[(rb + v) * 32 + col + 16] = (__bf16)s1[v];
    }
    asm volatile("s_wait_dscnt 0" ::: "memory");  // DS stores visible (same wave)
    const v16bf aP = load_a_frag(pT, 32);
    asm volatile("s_wait_dscnt 0" ::: "memory");  // DS loads done before next overwrite

    // O += P x V  (B^T fragments straight from Vt: contiguous along keys)
#pragma unroll
    for (int t = 0; t < 4; ++t) {
      v16bf bv = load_bt_frag(vbase + (size_t)(t * 16) * SEQ + jt * 32, SEQ);
      o[t] = wmma_bf16(aP, bv, o[t]);
    }
  }

  // normalize and write merged [B*S, D] in bf16
#pragma unroll
  for (int v = 0; v < 8; ++v) {
    const float inv = 1.0f / l_i[v];
    const size_t rowoff =
        ((size_t)(b * SEQ + q0 + rb + v)) * D_MODEL + h * HEAD_DIM;
#pragma unroll
    for (int t = 0; t < 4; ++t)
      merged[rowoff + t * 16 + col] = (__bf16)(o[t][v] * inv);
  }
}

// ---------------------------------------------------------------------------
// Host-side launcher
// ---------------------------------------------------------------------------
extern "C" void kernel_launch(void* const* d_in, const int* in_sizes, int n_in,
                              void* d_out, int out_size, void* d_ws, size_t ws_size,
                              hipStream_t stream) {
  const float* x    = (const float*)d_in[0];
  const float* W_q  = (const float*)d_in[1];
  const float* W_k  = (const float*)d_in[2];
  const float* W_v  = (const float*)d_in[3];
  const float* W_o  = (const float*)d_in[4];

  char* ws = (char*)d_ws;
  const size_t xb_sz   = (size_t)M_TOK * D_MODEL * sizeof(__bf16);   // 16 MB
  const size_t w_sz    = (size_t)D_MODEL * D_MODEL * sizeof(__bf16); // 2 MB

  __bf16* xb     = (__bf16*)(ws);
  __bf16* wq_t   = (__bf16*)(ws + xb_sz);
  __bf16* wk_t   = (__bf16*)(ws + xb_sz + 1 * w_sz);
  __bf16* wv_t   = (__bf16*)(ws + xb_sz + 2 * w_sz);
  __bf16* wo_t   = (__bf16*)(ws + xb_sz + 3 * w_sz);
  __bf16* Qb     = (__bf16*)(ws + xb_sz + 4 * w_sz);
  __bf16* Kb     = (__bf16*)(ws + 2 * xb_sz + 4 * w_sz);
  __bf16* Vtb    = (__bf16*)(ws + 3 * xb_sz + 4 * w_sz);
  __bf16* merged = (__bf16*)(ws + 4 * xb_sz + 4 * w_sz);

  // 1) precision conversion + weight transposes
  {
    const int n = M_TOK * D_MODEL;  // 8388608
    cvt_f32_to_bf16<<<(n + 255) / 256, 256, 0, stream>>>(x, xb, n);
    const int wthreads = D_MODEL * D_MODEL;  // 1048576
    transpose_w_bf16<<<wthreads / 256, 256, 0, stream>>>(W_q, wq_t);
    transpose_w_bf16<<<wthreads / 256, 256, 0, stream>>>(W_k, wk_t);
    transpose_w_bf16<<<wthreads / 256, 256, 0, stream>>>(W_v, wv_t);
    transpose_w_bf16<<<wthreads / 256, 256, 0, stream>>>(W_o, wo_t);
  }

  // 2) QKV projections (Q,K row-major bf16; V transposed per head)
  dim3 gGrid(D_MODEL / 64, M_TOK / 128);  // (16, 64)
  gemm_wmma_bf16<0><<<gGrid, 128, 0, stream>>>(xb, wq_t, Qb, M_TOK, D_MODEL, D_MODEL);
  gemm_wmma_bf16<0><<<gGrid, 128, 0, stream>>>(xb, wk_t, Kb, M_TOK, D_MODEL, D_MODEL);
  gemm_wmma_bf16<1><<<gGrid, 128, 0, stream>>>(xb, wv_t, Vtb, M_TOK, D_MODEL, D_MODEL);

  // 3) fused attention -> merged
  {
    const int waves = BATCH * HEADS * (SEQ / 16);  // 8192
    attn_wmma<<<waves / 4, 128, 0, stream>>>(Qb, Kb, Vtb, merged);
  }

  // 4) output projection -> f32 d_out
  gemm_wmma_bf16<2><<<gGrid, 128, 0, stream>>>(merged, wo_t, d_out, M_TOK, D_MODEL,
                                               D_MODEL);
}
